// Attention_68453188764029
// MI455X (gfx1250) — matrix operand order repaired and checked by reading, not compiled, not run on hardware
//
#include <hip/hip_runtime.h>
#include <hip/hip_bf16.h>

// ---------------------------------------------------------------------------
// Fused single-head attention for MI455X (gfx1250, wave32, WMMA).
//   pack:  W* -> bf16 in WMMA B-layout (one-time, 384 KB)
//   proj:  q = (Xq@Wq+bq)*(log2e/sqrt(D)) ; k = Xk@Wk+bk ; vT = (Xv@Wv+bv)^T
//   attn:  out = softmax2(mask(q k^T)) v   (flash-style, base-2 exp, f32 acc)
// ---------------------------------------------------------------------------

#define Bdim 4
#define Sdim 4096
#define Hdim 1024
#define Ddim 64
// 1/sqrt(64) * log2(e): scores move to the base-2 exponent domain
#define QSCALE (0.125f * 1.4426950408889634f)

typedef unsigned short u16;
typedef __attribute__((ext_vector_type(16))) __bf16 v16bf;
typedef __attribute__((ext_vector_type(8)))  float  v8f;

union BF16x16 {
    v16bf  v;
    __bf16 h[16];
    u16    s[16];
    uint4  u[2];
};
union F16x1 { __bf16 h; u16 s; };
union F4x4  { float4 v4[4]; float f[16]; };

// f32 -> bf16 via native conversion (backend emits v_cvt_pk_bf16_f32)
static __device__ __forceinline__ u16 f2bf(float f) {
    F16x1 cv; cv.h = (__bf16)f; return cv.s;
}

// ---------------------------------------------------------------------------
// Kernel 0: pack W into bf16 WMMA B-layout.
// Layout: wpack[chunk(32)][ntile(4)][lane(32)][e(16)]  (bf16)
//   lane = column (ntile*16 + lane&15), lane-half h holds K 8h..8h+7, 16+8h..16+8h+7.
// grid = 32, block = 128 (wave = ntile).  One launch per matrix.
// ---------------------------------------------------------------------------
__global__ __launch_bounds__(128) void pack_w_kernel(
    const float* __restrict__ W, u16* __restrict__ wpack)
{
    const int lane  = threadIdx.x & 31;
    const int ntile = threadIdx.x >> 5;
    const int chunk = blockIdx.x;

    const int l15   = lane & 15;
    const int bcol  = ntile * 16 + l15;
    const int kbase = chunk * 32 + (lane >> 4) * 8;
    u16* dst = wpack + (((long)chunk * 4 + ntile) * 32 + lane) * 16;
#pragma unroll
    for (int e = 0; e < 16; ++e)
        dst[e] = f2bf(W[(long)(kbase + e + (e & 8)) * Ddim + bcol]);
}

// ---------------------------------------------------------------------------
// Kernel 1: projection, templated on WHICH (0=Q scaled, 1=K, 2=V transposed).
// One wave = 16 rows x full D=64 (activations read from HBM exactly once).
// grid = B*S/16/8 = 128; block = 256 (8 waves).
// ---------------------------------------------------------------------------
template <int WHICH>
__global__ __launch_bounds__(256) void proj_kernel(
    const float* __restrict__ X, const u16* __restrict__ wpack,
    const float* __restrict__ bias, u16* __restrict__ dst)
{
    const int lane  = threadIdx.x & 31;
    const int wave  = threadIdx.x >> 5;
    const int rbase = (blockIdx.x * 8 + wave) * 16;         // 16 rows per wave
    const int halfw = lane >> 4;
    const int l15   = lane & 15;

    const long arowoff = (long)(rbase + l15) * Hdim;
    const int  acoff   = halfw * 8;                         // A-layout K sub-offset

    v8f acc[4];
    const v8f vzero = {0.f,0.f,0.f,0.f,0.f,0.f,0.f,0.f};
#pragma unroll
    for (int dt = 0; dt < 4; ++dt) acc[dt] = vzero;

    for (int chunk = 0; chunk < 32; ++chunk) {
        // A tile: 16 f32 -> bf16, two 32B spans (K c..c+7, c+16..c+23)
        F4x4 t;
        const float4* ap4 = (const float4*)(X + arowoff + chunk * 32 + acoff);
        t.v4[0] = ap4[0]; t.v4[1] = ap4[1];   // K c .. c+7
        t.v4[2] = ap4[4]; t.v4[3] = ap4[5];   // K c+16 .. c+23
        BF16x16 a;
#pragma unroll
        for (int j = 0; j < 16; ++j) a.h[j] = (__bf16)t.f[j];

        // 4 output tiles share the A operand
#pragma unroll
        for (int dt = 0; dt < 4; ++dt) {
            BF16x16 bm;
            const uint4* bp = (const uint4*)(wpack + (((long)chunk * 4 + dt) * 32 + lane) * 16);
            bm.u[0] = bp[0]; bm.u[1] = bp[1];
            acc[dt] = __builtin_amdgcn_wmma_f32_16x16x32_bf16(
                          false, a.v, false, bm.v, (short)0, acc[dt], false, false);
        }
    }

    // Epilogue: C layout — VGPR r: lanes 0..15 = row r, lanes 16..31 = row r+8
#pragma unroll
    for (int dt = 0; dt < 4; ++dt) {
        const int   col = dt * 16 + l15;
        const float bn  = bias[col];
#pragma unroll
        for (int r = 0; r < 8; ++r) {
            const int row = rbase + r + halfw * 8;
            if (WHICH == 0) {
                dst[(long)row * Ddim + col] = f2bf((acc[dt][r] + bn) * QSCALE);
            } else if (WHICH == 1) {
                dst[(long)row * Ddim + col] = f2bf(acc[dt][r] + bn);
            } else {
                const int bb = row >> 12;
                const int ss = row & (Sdim - 1);
                dst[((long)bb * Ddim + col) * Sdim + ss] = f2bf(acc[dt][r] + bn);
            }
        }
    }
}

// ---------------------------------------------------------------------------
// Kernel 2: flash attention.  One wave per 16-query tile, 32 keys / iteration.
// grid = B*(S/16)/8 = 128 blocks; block = 256 (8 waves).
// ---------------------------------------------------------------------------
__global__ __launch_bounds__(256) void attn_kernel(
    const u16* __restrict__ qb, const u16* __restrict__ kb,
    const u16* __restrict__ vT, const int* __restrict__ mask,
    float* __restrict__ out)
{
    __shared__ u16 ldsP[8][16 * 32];   // per-wave P bounce (C->A re-swizzle)

    const int lane  = threadIdx.x & 31;
    const int wave  = threadIdx.x >> 5;
    const int tile  = blockIdx.x * 8 + wave;   // 0..1023
    const int b     = tile >> 8;
    const int qbase = (tile & 255) * 16;
    const int halfw = lane >> 4;
    const int l15   = lane & 15;

    // Q tile (pre-scaled) in A layout, two K(=d) chunks of 32
    BF16x16 aq0, aq1;
    {
        const uint4* q4 = (const uint4*)(qb +
            (long)(b * Sdim + qbase + l15) * Ddim + halfw * 16);
        aq0.u[0] = q4[0]; aq0.u[1] = q4[1];
        aq1.u[0] = q4[4]; aq1.u[1] = q4[5];
    }

    const v8f vzero = {0.f,0.f,0.f,0.f,0.f,0.f,0.f,0.f};
    v8f acc[4]; acc[0] = vzero; acc[1] = vzero; acc[2] = vzero; acc[3] = vzero;
    float m[8], lp[8];                 // running max / per-lane partial sum
#pragma unroll
    for (int r = 0; r < 8; ++r) { m[r] = -3.0e38f; lp[r] = 0.f; }

    const u16* kbat  = kb + (long)b * Sdim * Ddim;
    const u16* vbat  = vT + (long)b * Ddim * Sdim;
    const int* maskp = mask + b * Sdim;

    for (int kt = 0; kt < Sdim / 32; ++kt) {
        const int kbase0 = kt * 32;
        const int kp0 = kbase0 + l15;
        const int kp1 = kbase0 + 16 + l15;

        // K B-tiles: lane = key column, lane-half splits d 0..15/16..31
        BF16x16 bk0lo, bk0hi, bk1lo, bk1hi;
        {
            const uint4* p0 = (const uint4*)(kbat + (long)kp0 * Ddim + halfw * 16);
            bk0lo.u[0] = p0[0]; bk0lo.u[1] = p0[1];
            bk0hi.u[0] = p0[4]; bk0hi.u[1] = p0[5];
            const uint4* p1 = (const uint4*)(kbat + (long)kp1 * Ddim + halfw * 16);
            bk1lo.u[0] = p1[0]; bk1lo.u[1] = p1[1];
            bk1hi.u[0] = p1[4]; bk1hi.u[1] = p1[5];
        }
        __builtin_prefetch(kbat + (long)(kp0 + 32) * Ddim, 0, 1);
        __builtin_prefetch(vbat + (long)l15 * Sdim + kbase0 + 32, 0, 1);

        // scores (already in base-2 exponent domain via Q pre-scale)
        v8f s0 = __builtin_amdgcn_wmma_f32_16x16x32_bf16(
                     false, aq0.v, false, bk0lo.v, (short)0, vzero, false, false);
        s0     = __builtin_amdgcn_wmma_f32_16x16x32_bf16(
                     false, aq1.v, false, bk0hi.v, (short)0, s0,    false, false);
        v8f s1 = __builtin_amdgcn_wmma_f32_16x16x32_bf16(
                     false, aq0.v, false, bk1lo.v, (short)0, vzero, false, false);
        s1     = __builtin_amdgcn_wmma_f32_16x16x32_bf16(
                     false, aq1.v, false, bk1hi.v, (short)0, s1,    false, false);

        // key mask as additive bias (one select per subtile)
        const float bias0 = maskp[kp0] ? 0.0f : -1.0e9f;
        const float bias1 = maskp[kp1] ? 0.0f : -1.0e9f;

        // online softmax: row-max shfl reduction only; row-sum kept per-lane
#pragma unroll
        for (int r = 0; r < 8; ++r) {
            const float sc0 = s0[r] + bias0;
            const float sc1 = s1[r] + bias1;
            float t = fmaxf(sc0, sc1);
            t = fmaxf(t, __shfl_xor(t, 1, 32));
            t = fmaxf(t, __shfl_xor(t, 2, 32));
            t = fmaxf(t, __shfl_xor(t, 4, 32));
            t = fmaxf(t, __shfl_xor(t, 8, 32));
            const float mn   = fmaxf(m[r], t);
            const float corr = __builtin_amdgcn_exp2f(m[r] - mn);
            m[r] = mn;
            const float p0 = __builtin_amdgcn_exp2f(sc0 - mn);
            const float p1 = __builtin_amdgcn_exp2f(sc1 - mn);
            lp[r] = lp[r] * corr + (p0 + p1);
#pragma unroll
            for (int dt = 0; dt < 4; ++dt) acc[dt][r] *= corr;

            const int row = r + halfw * 8;
            ldsP[wave][row * 32 + l15]      = f2bf(p0);
            ldsP[wave][row * 32 + 16 + l15] = f2bf(p1);
        }

        // per-wave in-order DS + explicit wait; clobber stops compiler reordering
        __asm volatile("s_wait_dscnt 0" ::: "memory");

        // P back in A layout (16x32, K = 32 keys)
        BF16x16 ap;
        {
            const uint4* pl = (const uint4*)&ldsP[wave][l15 * 32 + halfw * 16];
            ap.u[0] = pl[0];
            ap.u[1] = pl[1];
        }

        // out += P V : B-tile from transposed V, contiguous 32B per lane
        const u16* vrow = vbat + kbase0 + halfw * 16;
#pragma unroll
        for (int dt = 0; dt < 4; ++dt) {
            BF16x16 bvm;
            const uint4* pv = (const uint4*)(vrow + (long)(dt * 16 + l15) * Sdim);
            bvm.u[0] = pv[0]; bvm.u[1] = pv[1];
            acc[dt] = __builtin_amdgcn_wmma_f32_16x16x32_bf16(
                          false, ap.v, false, bvm.v, (short)0, acc[dt], false, false);
        }
    }

    // epilogue: reduce per-lane partial sums across the 16-lane half, normalize
    float rl[8];
#pragma unroll
    for (int r = 0; r < 8; ++r) {
        float rs = lp[r];
        rs += __shfl_xor(rs, 1, 32);
        rs += __shfl_xor(rs, 2, 32);
        rs += __shfl_xor(rs, 4, 32);
        rs += __shfl_xor(rs, 8, 32);
        rl[r] = 1.0f / rs;
    }
    float* outp = out + (long)(b * Sdim + qbase) * Ddim;
#pragma unroll
    for (int dt = 0; dt < 4; ++dt) {
#pragma unroll
        for (int r = 0; r < 8; ++r) {
            const int row = r + halfw * 8;
            outp[(long)row * Ddim + dt * 16 + l15] = acc[dt][r] * rl[r];
        }
    }
}

// ---------------------------------------------------------------------------
extern "C" void kernel_launch(void* const* d_in, const int* in_sizes, int n_in,
                              void* d_out, int out_size, void* d_ws, size_t ws_size,
                              hipStream_t stream) {
    const float* query = (const float*)d_in[0];
    const float* keyx  = (const float*)d_in[1];
    const float* value = (const float*)d_in[2];
    const int*   mask  = (const int*)  d_in[3];
    const float* Wq = (const float*)d_in[4];  const float* bq = (const float*)d_in[5];
    const float* Wk = (const float*)d_in[6];  const float* bk = (const float*)d_in[7];
    const float* Wv = (const float*)d_in[8];  const float* bv = (const float*)d_in[9];
    float* out = (float*)d_out;

    // workspace: q,k [B,S,D] bf16; vT [B,D,S] bf16; packed weights 3x128KB
    const size_t mat   = (size_t)Bdim * Sdim * Ddim;
    const size_t wmat  = (size_t)32 * 4 * 32 * 16;     // one packed W matrix
    u16* qb    = (u16*)d_ws;
    u16* kbp   = qb + mat;
    u16* vT    = kbp + mat;
    u16* wpq   = vT + mat;
    u16* wpk   = wpq + wmat;
    u16* wpv   = wpk + wmat;

    pack_w_kernel<<<32, 128, 0, stream>>>(Wq, wpq);
    pack_w_kernel<<<32, 128, 0, stream>>>(Wk, wpk);
    pack_w_kernel<<<32, 128, 0, stream>>>(Wv, wpv);

    const int gproj = (Bdim * Sdim) / 16 / 8;          // 128 blocks
    proj_kernel<0><<<gproj, 256, 0, stream>>>(query, wpq, bq, qb);
    proj_kernel<1><<<gproj, 256, 0, stream>>>(keyx,  wpk, bk, kbp);
    proj_kernel<2><<<gproj, 256, 0, stream>>>(value, wpv, bv, vT);

    const int qtiles = Bdim * (Sdim / 16);
    attn_kernel<<<qtiles / 8, 256, 0, stream>>>(qb, kbp, vT, mask, out);
}
